// RNN_32109175505717
// MI455X (gfx1250) — matrix operand order repaired and checked by reading, not compiled, not run on hardware
//
#include <hip/hip_runtime.h>
#include <cmath>

// ---------------------------------------------------------------------------
// Fused 3-layer tanh-RNN + FC head for MI455X (gfx1250, wave32, WMMA).
//
// All three recurrences advance together one time step at a time, so the
// inter-layer activation sequences (2 x 134 MB fp32 through HBM in the naive
// plan) never exist.  Per-workgroup state: 3 ping-ponged h tiles (16x64 f16)
// plus a zero-padded staged x tile, ~15 KB LDS.  Matrix work uses
// v_wmma_f32_16x16x32_f16 with fp32 accumulation; activations use the native
// gfx1250 v_tanh_f32 transcendental (branchless, 1 op).  A-fragment LDS loads
// are issued ahead of each layer's WMMA chain so DS latency overlaps the
// matrix pipeline.
// ---------------------------------------------------------------------------

typedef __attribute__((ext_vector_type(16))) _Float16 v16h;
typedef __attribute__((ext_vector_type(8)))  _Float16 v8h;
typedef __attribute__((ext_vector_type(2)))  _Float16 v2h;
typedef __attribute__((ext_vector_type(8)))  float    v8f;

#define T_LEN   128
#define F_IN    14
#define HID     64
#define BT      16      // batch rows per workgroup (one WMMA M tile)
#define NTHREADS 128    // 4 waves: one per 16-wide hidden slice

#define WMMA_F16(a, b, c) \
  __builtin_amdgcn_wmma_f32_16x16x32_f16(false, (a), false, (b), (short)0, (c), false, false)

// Native gfx1250 tanh (TRANS32 op) - branchless, on the recurrence's
// critical path 3*128 times per wave.
__device__ __forceinline__ float fast_tanh(float x) {
#if __has_builtin(__builtin_amdgcn_tanhf)
  return __builtin_amdgcn_tanhf(x);
#elif __has_builtin(__builtin_amdgcn_tanh_f32)
  return __builtin_amdgcn_tanh_f32(x);
#else
  float r;
  asm("v_tanh_f32 %0, %1" : "=v"(r) : "v"(x));
  return r;
#endif
}

// A fragment (16x32 f16) from an LDS row-major [16][stride] tile.
// Lane l holds row M = l&15; element i -> K = K0 + (l>>4)*16 + i
// => two contiguous 16B chunks per lane (ds_load_b128 x2).
__device__ __forceinline__ v16h load_a_frag(const _Float16* base, int stride,
                                            int K0, int lane) {
  const int rowM = lane & 15;
  const int kh   = (lane >> 4) * 16;
  const _Float16* p = base + rowM * stride + K0 + kh;
  v8h lo = *(const v8h*)(p);
  v8h hi = *(const v8h*)(p + 8);
  v16h a;
#pragma unroll
  for (int i = 0; i < 8; ++i) { a[i] = lo[i]; a[i + 8] = hi[i]; }
  return a;
}

// B fragment (32x16 f16) for out = in @ W^T  =>  B[k][n] = W[nRow][k].
// Element i of lane l: K = kBase + (l>>4)*16 + i, N = l&15.
// Per lane this is 16 contiguous floats of one W row; zero outside kLimit.
__device__ __forceinline__ v16h load_b_wT(const float* __restrict__ W, int nRow,
                                          int ncols, int kBase, int kLimit,
                                          int lane, bool valid) {
  const int k0 = kBase + (lane >> 4) * 16;
  const float* p = W + (size_t)(valid ? nRow : 0) * ncols + k0;
  v16h b;
#pragma unroll
  for (int i = 0; i < 16; ++i) {
    float w = (valid && (k0 + i) < kLimit) ? p[i] : 0.0f;
    b[i] = (_Float16)w;
  }
  return b;
}

__global__ void __launch_bounds__(NTHREADS)
rnn3_fused_wmma(const float* __restrict__ x,
                const float* __restrict__ Wih0, const float* __restrict__ Whh0,
                const float* __restrict__ bih0, const float* __restrict__ bhh0,
                const float* __restrict__ Wih1, const float* __restrict__ Whh1,
                const float* __restrict__ bih1, const float* __restrict__ bhh1,
                const float* __restrict__ Wih2, const float* __restrict__ Whh2,
                const float* __restrict__ bih2, const float* __restrict__ bhh2,
                const float* __restrict__ fc1w, const float* __restrict__ fc1b,
                const float* __restrict__ fc2w, const float* __restrict__ fc2b,
                float* __restrict__ out) {
  __shared__ alignas(16) _Float16 sH[3][2][BT][HID];  // 3 layers, ping-pong
  __shared__ alignas(16) _Float16 sX[2][BT][32];      // x tile, K padded 14->32
  __shared__ alignas(16) _Float16 sY[BT][32];         // fc1 output

  const int tid      = threadIdx.x;
  const int wave     = tid >> 5;
  const int lane     = tid & 31;
  const int laneN    = lane & 15;
  const int laneHalf = lane >> 4;
  const int nBase    = wave * 16;            // hidden-slice base for this wave
  const int bBase    = blockIdx.x * BT;      // batch-row base for this WG

  // x staging assignment: 112 threads each own one (row, float2) of the
  // 16x14 real x tile -> one aligned 8B global load + one 4B LDS store.
  const bool xload = (tid < (BT * F_IN / 2));          // tid < 112
  const int  xr    = tid / (F_IN / 2);                 // 0..15  (row)
  const int  xk    = (tid % (F_IN / 2)) * 2;           // 0,2,...,12
  const float* xrow = x + ((size_t)(bBase + xr) * T_LEN) * F_IN + xk;

  // ---- loop-invariant weight B-fragments, register resident -------------
  const int  nRow   = nBase + laneN;         // 0..63
  const bool validF = (nRow < 32);

  v16h bX0  = load_b_wT(Wih0, nRow, F_IN, 0,  F_IN, lane, true);   // x proj (K pad)
  v16h bH00 = load_b_wT(Whh0, nRow, HID, 0,  HID, lane, true);
  v16h bH01 = load_b_wT(Whh0, nRow, HID, 32, HID, lane, true);
  v16h bI10 = load_b_wT(Wih1, nRow, HID, 0,  HID, lane, true);
  v16h bI11 = load_b_wT(Wih1, nRow, HID, 32, HID, lane, true);
  v16h bH10 = load_b_wT(Whh1, nRow, HID, 0,  HID, lane, true);
  v16h bH11 = load_b_wT(Whh1, nRow, HID, 32, HID, lane, true);
  v16h bI20 = load_b_wT(Wih2, nRow, HID, 0,  HID, lane, true);
  v16h bI21 = load_b_wT(Wih2, nRow, HID, 32, HID, lane, true);
  v16h bH20 = load_b_wT(Whh2, nRow, HID, 0,  HID, lane, true);
  v16h bH21 = load_b_wT(Whh2, nRow, HID, 32, HID, lane, true);
  v16h bF10 = load_b_wT(fc1w, nRow, HID, 0,  HID, lane, validF);
  v16h bF11 = load_b_wT(fc1w, nRow, HID, 32, HID, lane, validF);

  const float bias0 = bih0[nRow] + bhh0[nRow];
  const float bias1 = bih1[nRow] + bhh1[nRow];
  const float bias2 = bih2[nRow] + bhh2[nRow];
  const float biasF = validF ? fc1b[nRow] : 0.0f;

  // ---- zero h state (both buffers) and both x buffers (incl. K pad) -----
  for (int i = tid; i < 3 * 2 * BT * HID; i += NTHREADS)
    ((_Float16*)sH)[i] = (_Float16)0.0f;
  for (int i = tid; i < 2 * BT * 32; i += NTHREADS)
    ((_Float16*)sX)[i] = (_Float16)0.0f;

  // ---- stage x(t=0) real elements (pad stays zero forever) --------------
  if (xload) {
    const float2 v = *(const float2*)(xrow);           // t = 0
    v2h p; p[0] = (_Float16)v.x; p[1] = (_Float16)v.y;
    *(v2h*)&sX[0][xr][xk] = p;
  }
  __syncthreads();

  // ---- main recurrence: all 3 layers advance per step -------------------
#pragma unroll 1
  for (int t = 0; t < T_LEN; ++t) {
    const int rp = t & 1, wp = rp ^ 1;

    // Prefetch x(t+1) into registers; stored to LDS at end of step.
    // (t+1)&127 wraps at t=127 to a valid address; that value is dead.
    float2 xv = {0.0f, 0.0f};
    if (xload) {
      const int tn = (t + 1) & (T_LEN - 1);
      xv = *(const float2*)(xrow + (size_t)tn * F_IN);
    }

    // ---- layer 0: h0 = tanh(x_t @ Wih0^T + h0 @ Whh0^T + b) ----
    {
      // issue all A-fragment DS loads up front, then run the WMMA chain
      v16h a0 = load_a_frag(&sX[rp][0][0], 32, 0, lane);
      v16h a1 = load_a_frag(&sH[0][rp][0][0], HID, 0, lane);
      v16h a2 = load_a_frag(&sH[0][rp][0][0], HID, 32, lane);
      v8f acc = {};
      acc = WMMA_F16(a0, bX0,  acc);
      acc = WMMA_F16(a1, bH00, acc);
      acc = WMMA_F16(a2, bH01, acc);
#pragma unroll
      for (int r = 0; r < 8; ++r) {
        float h = fast_tanh(acc[r] + bias0);
        sH[0][wp][r + laneHalf * 8][nBase + laneN] = (_Float16)h;
      }
    }
    __syncthreads();

    // ---- layer 1: h1 = tanh(h0 @ Wih1^T + h1 @ Whh1^T + b) ----
    {
      v16h a0 = load_a_frag(&sH[0][wp][0][0], HID, 0, lane);
      v16h a1 = load_a_frag(&sH[0][wp][0][0], HID, 32, lane);
      v16h a2 = load_a_frag(&sH[1][rp][0][0], HID, 0, lane);
      v16h a3 = load_a_frag(&sH[1][rp][0][0], HID, 32, lane);
      v8f acc = {};
      acc = WMMA_F16(a0, bI10, acc);
      acc = WMMA_F16(a1, bI11, acc);
      acc = WMMA_F16(a2, bH10, acc);
      acc = WMMA_F16(a3, bH11, acc);
#pragma unroll
      for (int r = 0; r < 8; ++r) {
        float h = fast_tanh(acc[r] + bias1);
        sH[1][wp][r + laneHalf * 8][nBase + laneN] = (_Float16)h;
      }
    }
    __syncthreads();

    // ---- layer 2: h2 = tanh(h1 @ Wih2^T + h2 @ Whh2^T + b) ----
    {
      v16h a0 = load_a_frag(&sH[1][wp][0][0], HID, 0, lane);
      v16h a1 = load_a_frag(&sH[1][wp][0][0], HID, 32, lane);
      v16h a2 = load_a_frag(&sH[2][rp][0][0], HID, 0, lane);
      v16h a3 = load_a_frag(&sH[2][rp][0][0], HID, 32, lane);
      v8f acc = {};
      acc = WMMA_F16(a0, bI20, acc);
      acc = WMMA_F16(a1, bI21, acc);
      acc = WMMA_F16(a2, bH20, acc);
      acc = WMMA_F16(a3, bH21, acc);
#pragma unroll
      for (int r = 0; r < 8; ++r) {
        float h = fast_tanh(acc[r] + bias2);
        sH[2][wp][r + laneHalf * 8][nBase + laneN] = (_Float16)h;
      }
    }

    // stage prefetched x(t+1) into the other buffer (pad untouched)
    if (xload) {
      v2h p; p[0] = (_Float16)xv.x; p[1] = (_Float16)xv.y;
      *(v2h*)&sX[wp][xr][xk] = p;
    }
    __syncthreads();
  }

  // ---- head: y = relu(h2_last @ fc1^T + b1); out = y @ fc2^T + b2 ------
  const int fb = T_LEN & 1;   // buffer holding h at t = T-1 (wp of last step)
  {
    v16h a0 = load_a_frag(&sH[2][fb][0][0], HID, 0, lane);
    v16h a1 = load_a_frag(&sH[2][fb][0][0], HID, 32, lane);
    v8f acc = {};
    acc = WMMA_F16(a0, bF10, acc);
    acc = WMMA_F16(a1, bF11, acc);
    if (validF) {   // waves 0,1 cover the 32 fc1 outputs
#pragma unroll
      for (int r = 0; r < 8; ++r) {
        float y = acc[r] + biasF;
        y = (y > 0.0f) ? y : 0.0f;
        sY[r + laneHalf * 8][nRow] = (_Float16)y;
      }
    }
  }
  __syncthreads();

  if (tid < BT) {
    float s = fc2b[0];
#pragma unroll
    for (int j = 0; j < 32; ++j) s += (float)sY[tid][j] * fc2w[j];
    out[bBase + tid] = s;
  }
}

extern "C" void kernel_launch(void* const* d_in, const int* in_sizes, int n_in,
                              void* d_out, int out_size, void* d_ws, size_t ws_size,
                              hipStream_t stream) {
  (void)n_in; (void)d_ws; (void)ws_size; (void)out_size;
  const float* xp   = (const float*)d_in[0];
  const float* Wih0 = (const float*)d_in[1];
  const float* Whh0 = (const float*)d_in[2];
  const float* bih0 = (const float*)d_in[3];
  const float* bhh0 = (const float*)d_in[4];
  const float* Wih1 = (const float*)d_in[5];
  const float* Whh1 = (const float*)d_in[6];
  const float* bih1 = (const float*)d_in[7];
  const float* bhh1 = (const float*)d_in[8];
  const float* Wih2 = (const float*)d_in[9];
  const float* Whh2 = (const float*)d_in[10];
  const float* bih2 = (const float*)d_in[11];
  const float* bhh2 = (const float*)d_in[12];
  const float* fc1w = (const float*)d_in[13];
  const float* fc1b = (const float*)d_in[14];
  const float* fc2w = (const float*)d_in[15];
  const float* fc2b = (const float*)d_in[16];
  float* out = (float*)d_out;

  const int B = in_sizes[0] / (T_LEN * F_IN);   // 4096
  const int grid = B / BT;                      // 256 workgroups

  rnn3_fused_wmma<<<grid, NTHREADS, 0, stream>>>(
      xp, Wih0, Whh0, bih0, bhh0, Wih1, Whh1, bih1, bhh1,
      Wih2, Whh2, bih2, bhh2, fc1w, fc1b, fc2w, fc2b, out);
}
